// SE3PointConvolution_59940563583479
// MI455X (gfx1250) — matrix operand order repaired and checked by reading, not compiled, not run on hardware
//
#include <hip/hip_runtime.h>
#include <hip/hip_bf16.h>

typedef __attribute__((ext_vector_type(16))) __bf16 v16bf;
typedef __attribute__((ext_vector_type(8)))  __bf16 v8bf;
typedef __attribute__((ext_vector_type(8)))  float  v8f;

#define NEG_GAMMA_LOG2E (-8.8365071254949257f)   // -6.125 * log2(e)

// ---------------------------------------------------------------------------
// Kernel 1: T[n][d][a*8+r] = sum_c input[n][c][a] * W[d][c][r], stored bf16.
// grid = 64 blocks (n*16+d), 256 threads (a). W[d] is block-uniform -> s_loads.
// Each thread emits one 16B store (8 contiguous bf16 over r).
// ---------------------------------------------------------------------------
__global__ __launch_bounds__(256) void se3_prep_T(
    const float* __restrict__ input,   // [4][16][256]
    const float* __restrict__ W,       // [16][16][8]
    __bf16* __restrict__ T)            // [4][16][2048]
{
  const int bd = blockIdx.x;
  const int d  = bd & 15;
  const int n  = bd >> 4;
  const int a  = threadIdx.x;

  float in_c[16];
#pragma unroll
  for (int c = 0; c < 16; ++c)
    in_c[c] = input[(n * 16 + c) * 256 + a];

  const float* wp = W + d * 128;       // [c][r], uniform per block
  v8bf tv;
#pragma unroll
  for (int r = 0; r < 8; ++r) {
    float s = 0.f;
#pragma unroll
    for (int c = 0; c < 16; ++c)
      s += in_c[c] * wp[c * 8 + r];
    tv[r] = (__bf16)s;
  }
  // flat index (n*16+d)*2048 + a*8
  *(v8bf*)(T + (size_t)bd * 2048 + a * 8) = tv;
}

// ---------------------------------------------------------------------------
// Kernel 2: out[n][d][b] = sum_{K=(a,r)} T[n][d][K] * R[n][b][K]
// One workgroup per (n, b-tile of 16). 4 waves split K=2048 -> 16 WMMAs each.
// A fragments loaded from precomputed bf16 T; B fragments (gaussian RBF * mask)
// synthesized in registers per lane; f32 partials reduced across waves in LDS.
// ---------------------------------------------------------------------------
__global__ __launch_bounds__(128) void se3_wmma(
    const __bf16* __restrict__ T,      // [4][16][2048]
    const float* __restrict__ diff,    // [4][256][256][3]
    const float* __restrict__ mask,    // [4][256][256]
    const float* __restrict__ radii,   // [8]
    float* __restrict__ out)           // [4][16][256]
{
  __shared__ float red[4][256];

  const int n     = blockIdx.x >> 4;
  const int btile = blockIdx.x & 15;
  const int tid   = threadIdx.x;
  const int wave  = tid >> 5;
  const int lane  = tid & 31;
  const int bl    = lane & 15;         // N column / A-matrix M row owner
  const int h     = lane >> 4;         // K-half selector
  const int b     = btile * 16 + bl;

  float rad[8];
#pragma unroll
  for (int i = 0; i < 8; ++i) rad[i] = radii[i];   // uniform -> scalar loads

  const float*  dbase = diff + ((size_t)(n * 256 + b)) * 256 * 3;
  const float*  mbase = mask + (size_t)(n * 256 + b) * 256;
  const __bf16* Abase = T + (size_t)(n * 16 + bl) * 2048;

  v8f acc = {};

  // wave handles k-blocks kb = wave*16 .. wave*16+15, K = kb*32 .. kb*32+31
#pragma unroll 1
  for (int kk = 0; kk < 16; ++kk) {
    const int kb = wave * 16 + kk;

    // ---- A fragment (contiguous k = 16h + s, same as B): lane m = bl, half h.
    // elements 0..7  <- K = kb*32 + h*16     .. +7   (VGPRs 0-3)
    // elements 8..15 <- K = kb*32 + h*16+8   .. +7   (VGPRs 4-7)
    v8bf a_lo = *(const v8bf*)(Abase + kb * 32 + h * 16);
    v8bf a_hi = *(const v8bf*)(Abase + kb * 32 + h * 16 + 8);
    v16bf avec;
#pragma unroll
    for (int j = 0; j < 8; ++j) { avec[j] = a_lo[j]; avec[8 + j] = a_hi[j]; }

    // ---- B fragment: lane owns N=bl, K = kb*32 + h*16 + [0..15]
    //      = 2 consecutive a-values x 8 radii (r minor in K)
    const int    a0 = kb * 4 + h * 2;
    const float* d0 = dbase + a0 * 3;
    __builtin_prefetch((const void*)(d0 + 12), 0, 0);   // next kb's diff line

    const float dx0 = d0[0], dy0 = d0[1], dz0 = d0[2];
    const float dx1 = d0[3], dy1 = d0[4], dz1 = d0[5];
    const float m0  = mbase[a0], m1 = mbase[a0 + 1];
    const float dist0 = __builtin_sqrtf(dx0*dx0 + dy0*dy0 + dz0*dz0);
    const float dist1 = __builtin_sqrtf(dx1*dx1 + dy1*dy1 + dz1*dz1);

    v16bf bvec;
#pragma unroll
    for (int r = 0; r < 8; ++r) {
      const float t0 = dist0 - rad[r];
      const float t1 = dist1 - rad[r];
      const float R0 = __builtin_amdgcn_exp2f(NEG_GAMMA_LOG2E * t0 * t0) * m0;
      const float R1 = __builtin_amdgcn_exp2f(NEG_GAMMA_LOG2E * t1 * t1) * m1;
      bvec[r]     = (__bf16)R0;
      bvec[8 + r] = (__bf16)R1;
    }

    acc = __builtin_amdgcn_wmma_f32_16x16x32_bf16(
        /*neg_a=*/false, avec, /*neg_b=*/false, bvec,
        /*c_mod=*/(short)0, acc, /*reuse_a=*/false, /*reuse_b=*/false);
  }

  // ---- cross-wave reduction of the 16x16 f32 tile through LDS
#pragma unroll
  for (int i = 0; i < 8; ++i) red[wave][i * 32 + lane] = acc[i];
  __syncthreads();

  if (wave == 0) {
#pragma unroll
    for (int i = 0; i < 8; ++i) {
      const float s = red[0][i * 32 + lane] + red[1][i * 32 + lane] +
                      red[2][i * 32 + lane] + red[3][i * 32 + lane];
      // C-fragment layout: VGPR i, lanes 0-15 -> M=i, lanes 16-31 -> M=i+8
      const int dd = i + 8 * h;
      out[(n * 16 + dd) * 256 + btile * 16 + bl] = s;
    }
  }
}

// ---------------------------------------------------------------------------
extern "C" void kernel_launch(void* const* d_in, const int* in_sizes, int n_in,
                              void* d_out, int out_size, void* d_ws, size_t ws_size,
                              hipStream_t stream) {
  const float* input = (const float*)d_in[0];   // [4][16][256]
  const float* diff  = (const float*)d_in[1];   // [4][256][256][3]
  const float* mask  = (const float*)d_in[2];   // [4][256][256]
  const float* W     = (const float*)d_in[3];   // [16][16][8]
  const float* radii = (const float*)d_in[4];   // [8]
  float* out = (float*)d_out;                   // [4][16][256]

  __bf16* T = (__bf16*)d_ws;                    // 131072 bf16 = 256 KB scratch

  se3_prep_T<<<64, 256, 0, stream>>>(input, W, T);
  se3_wmma<<<64, 128, 0, stream>>>(T, diff, mask, radii, out);
}
